// DeceptionNet_18013092840193
// MI455X (gfx1250) — matrix operand order repaired and checked by reading, not compiled, not run on hardware
//
#include <hip/hip_runtime.h>
#include <hip/hip_bf16.h>
#include <cstdint>
#include <cstddef>

#define B_   32
#define T_   2048
#define H_   128
#define NG_  512      // 4*H
#define EPS_ 1e-5f

typedef __attribute__((ext_vector_type(16))) _Float16 v16h;
typedef __attribute__((ext_vector_type(8)))  _Float16 v8h;
typedef __attribute__((ext_vector_type(8)))  float    v8f;

union U16h { v16h v; v8h h[2]; };
union U8f  { v8f v; float e[8]; };

__device__ __forceinline__ float sigf(float x) { return 1.0f / (1.0f + expf(-x)); }

// Async global->LDS copy (CDNA5 GLOBAL_LOAD_ASYNC_TO_LDS_B128, tracked by ASYNCcnt).
// lds_off = LDS byte offset (low 32 bits of a generic shared pointer).
__device__ __forceinline__ void async_ld_b128(uint32_t lds_off, const void* gaddr) {
    asm volatile("global_load_async_to_lds_b128 %0, %1, off"
                 :: "v"(lds_off), "v"((uint64_t)(uintptr_t)gaddr) : "memory");
}
__device__ __forceinline__ void wait_async0() {
    asm volatile("s_wait_asynccnt 0x0" ::: "memory");
}

// ---------------------------------------------------------------------------
// Generic WMMA GEMM:  Y[M,N] (f32) = A[M,K] (f16, row-major) @ W[N,K]^T (f16) + bias[N]
// One wave per 16x16 output tile. K multiple of 32, M,N multiples of 16.
// ---------------------------------------------------------------------------
__global__ __launch_bounds__(128) void k_gemm_bias(
    const _Float16* __restrict__ A, const _Float16* __restrict__ W,
    const float* __restrict__ bias, float* __restrict__ Y,
    int M, int N, int K)
{
    int wave = blockIdx.x * 4 + (threadIdx.x >> 5);
    int lane = threadIdx.x & 31;
    int tilesN = N >> 4;
    int tiles  = (M >> 4) * tilesN;
    if (wave >= tiles) return;                 // wave-uniform: EXEC stays all-ones
    int tm = wave / tilesN, tn = wave - tm * tilesN;
    int g = lane >> 4, l = lane & 15;

    const _Float16* Ar = A + (size_t)(tm * 16 + l) * K;   // A row m = tm*16 + (lane&15)
    const _Float16* Wr = W + (size_t)(tn * 16 + l) * K;   // B col n = tn*16 + (lane&15)

    U8f acc = {};
    for (int k0 = 0; k0 < K; k0 += 32) {
        U16h a, b;
        a.h[0] = *(const v8h*)(Ar + k0 +      g * 8);     // K = k0 + g*8  + [0..7]
        a.h[1] = *(const v8h*)(Ar + k0 + 16 + g * 8);     // K = k0+16+g*8 + [0..7]
        b.h[0] = *(const v8h*)(Wr + k0 +      g * 8);     // K = k0 + g*8  + [0..7]
        b.h[1] = *(const v8h*)(Wr + k0 + 16 + g * 8);     // K = k0+16+g*8 + [0..7]
        acc.v = __builtin_amdgcn_wmma_f32_16x16x32_f16(
                    false, a.v, false, b.v, (short)0, acc.v, false, false);
    }
    int ncol = tn * 16 + l;
    float bv = bias ? bias[ncol] : 0.0f;
    for (int r = 0; r < 8; ++r) {                         // C: VGPR r -> M = r + g*8
        int m = tm * 16 + g * 8 + r;
        Y[(size_t)m * N + ncol] = acc.e[r] + bv;
    }
}

// ---------------------------------------------------------------------------
// Persistent single-workgroup LSTM scan. 512 threads = 16 waves.
// LDS: h(f16 32x128) + c(f32 32x128) + gates/xg-stage(f32 32x512) + whh(f16 512x128) = 216KB.
// gS is double-duty: the async engine deposits xg[:,t,:] there; phase A accumulates
// the recurrent WMMA product into it; phase B consumes it; then xg[:,t+1,:] is
// async-staged into it UNDER the transcendental cell math (ASYNCcnt overlap).
// ---------------------------------------------------------------------------
__global__ __launch_bounds__(512) void k_lstm_scan(
    const float* __restrict__ xg,          // [B*T, 512], row = b*T + t (bias already added)
    const _Float16* __restrict__ whh,      // [512, 128] f16
    float* __restrict__ out,               // out[(b*T+t)*outStride + outOff + j]
    const int* __restrict__ lens,
    int outStride, int outOff, int applyMask)
{
    extern __shared__ char smem[];
    _Float16* hS = (_Float16*)smem;                          // 32*128*2  =   8192 B
    float*    cS = (float*)(smem + 8192);                    // 32*128*4  =  16384 B
    float*    gS = (float*)(smem + 8192 + 16384);            // 32*512*4  =  65536 B
    _Float16* wS = (_Float16*)(smem + 8192 + 16384 + 65536); // 512*128*2 = 131072 B

    int tid = threadIdx.x;
    uint32_t gS_lds = (uint32_t)(uintptr_t)gS;               // LDS byte offsets
    uint32_t wS_lds = (uint32_t)(uintptr_t)wS;

    // one-time async preload: whh (128KB) and xg[:,0,:] (64KB) -> LDS
    for (int i = tid; i < (NG_ * H_ * 2) / 16; i += 512)
        async_ld_b128(wS_lds + (uint32_t)i * 16, (const char*)whh + (size_t)i * 16);
    for (int i = tid; i < (B_ * NG_) / 4; i += 512) {
        int b = i >> 7, n0 = (i & 127) << 2;
        async_ld_b128(gS_lds + (uint32_t)i * 16, xg + ((size_t)b * T_) * NG_ + n0);
    }
    for (int i = tid; i < B_ * H_; i += 512) { hS[i] = (_Float16)0.0f; cS[i] = 0.0f; }
    wait_async0();
    __syncthreads();

    int wave = tid >> 5, lane = tid & 31;
    int g = lane >> 4, l = lane & 15;

    for (int t = 0; t < T_; ++t) {
        // -------- phase A: gS += h @ whh^T (64 tiles, 4 per wave) --------
        for (int q = 0; q < 4; ++q) {
            int tile = wave + (q << 4);
            int tm = tile >> 5, tn = tile & 31;
            int ncol = tn * 16 + l;
            const _Float16* hr = hS + (size_t)(tm * 16 + l) * H_;
            const _Float16* wr = wS + (size_t)ncol * H_;
            U8f acc = {};
            for (int k0 = 0; k0 < H_; k0 += 32) {
                U16h a, b;
                a.h[0] = *(const v8h*)(hr + k0 +      g * 8);
                a.h[1] = *(const v8h*)(hr + k0 + 16 + g * 8);
                b.h[0] = *(const v8h*)(wr + k0 +      g * 8);
                b.h[1] = *(const v8h*)(wr + k0 + 16 + g * 8);
                acc.v = __builtin_amdgcn_wmma_f32_16x16x32_f16(
                            false, a.v, false, b.v, (short)0, acc.v, false, false);
            }
            for (int r = 0; r < 8; ++r) {                    // unique owner per element
                int bb = tm * 16 + g * 8 + r;
                gS[bb * NG_ + ncol] += acc.e[r];
            }
        }
        __syncthreads();

        // -------- phase B1: gates -> registers, then gS is dead --------
        float gi[8], gf[8], gg[8], go[8];
        for (int p = 0; p < 8; ++p) {
            int idx = tid + (p << 9);
            int b = idx >> 7, j = idx & (H_ - 1);
            gi[p] = gS[b * NG_ + j];
            gf[p] = gS[b * NG_ + H_ + j];
            gg[p] = gS[b * NG_ + 2 * H_ + j];
            go[p] = gS[b * NG_ + 3 * H_ + j];
        }
        __syncthreads();

        // async-stage xg[:, t+1, :] into gS while doing the cell math below
        if (t + 1 < T_) {
            for (int i = tid; i < (B_ * NG_) / 4; i += 512) {
                int b = i >> 7, n0 = (i & 127) << 2;
                async_ld_b128(gS_lds + (uint32_t)i * 16,
                              xg + ((size_t)b * T_ + (t + 1)) * NG_ + n0);
            }
        }

        // -------- phase B2: LSTM cell (transcendental heavy) + store --------
        for (int p = 0; p < 8; ++p) {
            int idx = tid + (p << 9);
            int b = idx >> 7, j = idx & (H_ - 1);
            float c = sigf(gf[p]) * cS[idx] + sigf(gi[p]) * tanhf(gg[p]);
            float h = sigf(go[p]) * tanhf(c);
            cS[idx] = c;
            hS[idx] = (_Float16)h;
            float ov = h;
            if (applyMask && t >= lens[b]) ov = 0.0f;
            out[((size_t)b * T_ + t) * outStride + outOff + j] = ov;
        }
        if (t + 1 < T_) wait_async0();
        __syncthreads();
    }
}

// ---------------------------------------------------------------------------
// Conv1d (k=3, pad=1) + BN + ReLU over [B,T,C] layout.
// ---------------------------------------------------------------------------
__global__ void k_conv_bn_relu(const float* __restrict__ in, const float* __restrict__ w,
                               const float* __restrict__ bias,
                               const float* __restrict__ gm, const float* __restrict__ bt,
                               const float* __restrict__ mean, const float* __restrict__ var,
                               float* __restrict__ out, int Cin, int Cout)
{
    int idx = blockIdx.x * 256 + threadIdx.x;
    int total = B_ * T_ * Cout;
    if (idx >= total) return;
    int co = idx % Cout;
    int t  = (idx / Cout) % T_;
    int b  = idx / (Cout * T_);
    float acc = bias[co];
    for (int k = 0; k < 3; ++k) {
        int tt = t + k - 1;
        if (tt < 0 || tt >= T_) continue;
        const float* row = in + ((size_t)b * T_ + tt) * Cin;
        const float* wr  = w + ((size_t)co * Cin) * 3 + k;
        for (int ci = 0; ci < Cin; ++ci) acc += row[ci] * wr[(size_t)ci * 3];
    }
    float s = gm[co] * rsqrtf(var[co] + EPS_);
    acc = (acc - mean[co]) * s + bt[co];
    out[(size_t)idx] = fmaxf(acc, 0.0f);
}

// -------------------- casts / reversals --------------------
__global__ void k_cast_w16(const float* __restrict__ in, _Float16* __restrict__ out, int n) {
    int i = blockIdx.x * 256 + threadIdx.x;
    if (i < n) out[i] = (_Float16)in[i];
}

__global__ void k_cast_act(const float* __restrict__ in, _Float16* __restrict__ out,
                           const int* __restrict__ lens, int C, int useMask) {
    int i = blockIdx.x * 256 + threadIdx.x;
    int total = B_ * T_ * C;
    if (i >= total) return;
    int t = (i / C) % T_;
    int b = i / (C * T_);
    float v = in[i];
    if (useMask && t >= lens[b]) v = 0.0f;
    out[i] = (_Float16)v;
}

__global__ void k_revcast(const float* __restrict__ in, _Float16* __restrict__ out,
                          const int* __restrict__ lens, int C) {
    int i = blockIdx.x * 256 + threadIdx.x;
    int total = B_ * T_ * C;
    if (i >= total) return;
    int c = i % C;
    int t = (i / C) % T_;
    int b = i / (C * T_);
    int len = lens[b];
    _Float16 v = (_Float16)0.0f;
    if (t < len) v = (_Float16)in[((size_t)b * T_ + (len - 1 - t)) * C + c];
    out[i] = v;
}

__global__ void k_rev_store(const float* __restrict__ in, float* __restrict__ out,
                            const int* __restrict__ lens, int C, int outStride, int outOff) {
    int i = blockIdx.x * 256 + threadIdx.x;
    int total = B_ * T_ * C;
    if (i >= total) return;
    int c = i % C;
    int t = (i / C) % T_;
    int b = i / (C * T_);
    int len = lens[b];
    float v = 0.0f;
    if (t < len) v = in[((size_t)b * T_ + (len - 1 - t)) * C + c];
    out[((size_t)b * T_ + t) * outStride + outOff + c] = v;
}

// -------------------- LayerNorm over last dim (C=256), one wave32 per row ----
__global__ void k_layernorm(const float* __restrict__ in, const float* __restrict__ gm,
                            const float* __restrict__ bt, float* __restrict__ out, int C) {
    int wave = blockIdx.x * 8 + (threadIdx.x >> 5);
    int lane = threadIdx.x & 31;
    const float* row = in + (size_t)wave * C;
    float s = 0.0f, s2 = 0.0f;
    for (int i = lane; i < C; i += 32) { float v = row[i]; s += v; s2 += v * v; }
    for (int o = 16; o > 0; o >>= 1) { s += __shfl_xor(s, o, 32); s2 += __shfl_xor(s2, o, 32); }
    float mu = s / C;
    float inv = rsqrtf(s2 / C - mu * mu + EPS_);
    float* orow = out + (size_t)wave * C;
    for (int i = lane; i < C; i += 32) orow[i] = (row[i] - mu) * inv * gm[i] + bt[i];
}

// -------------------- attention (head 0 only; rest is dead in reference) -----
__global__ void k_scores(const float* __restrict__ t1, const float* __restrict__ w2,
                         const float* __restrict__ b2, float* __restrict__ scores) {
    int row = blockIdx.x * 256 + threadIdx.x;          // B*T rows
    float acc = 0.0f;
    const float* r = t1 + (size_t)row * 64;
    for (int k = 0; k < 64; ++k) acc += w2[k] * tanhf(r[k]);
    scores[row] = acc + b2[0];
}

__global__ void k_softmax_T(const float* __restrict__ scores, float* __restrict__ aw) {
    __shared__ float red[256];
    int b = blockIdx.x, tid = threadIdx.x;
    const float* s = scores + (size_t)b * T_;
    float m = -1e30f;
    for (int t = tid; t < T_; t += 256) m = fmaxf(m, s[t]);
    red[tid] = m; __syncthreads();
    for (int o = 128; o > 0; o >>= 1) { if (tid < o) red[tid] = fmaxf(red[tid], red[tid + o]); __syncthreads(); }
    m = red[0]; __syncthreads();
    float sum = 0.0f;
    for (int t = tid; t < T_; t += 256) sum += expf(s[t] - m);
    red[tid] = sum; __syncthreads();
    for (int o = 128; o > 0; o >>= 1) { if (tid < o) red[tid] += red[tid + o]; __syncthreads(); }
    float inv = 1.0f / red[0];
    for (int t = tid; t < T_; t += 256) aw[(size_t)b * T_ + t] = expf(s[t] - m) * inv;
}

__global__ void k_ctx(const float* __restrict__ aw, const float* __restrict__ nm,
                      float* __restrict__ z, int outStride) {
    int b = blockIdx.x, d = threadIdx.x;               // 256 threads
    float acc = 0.0f;
    for (int t = 0; t < T_; ++t)
        acc += aw[(size_t)b * T_ + t] * nm[((size_t)b * T_ + t) * 256 + d];
    z[(size_t)b * outStride + d] = acc;
}

// -------------------- global-feature MLP + residual LN -----------------------
__global__ void k_glob(const float* __restrict__ g, const float* __restrict__ gw1,
                       const float* __restrict__ gb1, const float* __restrict__ gw2,
                       const float* __restrict__ gb2, const float* __restrict__ glg,
                       const float* __restrict__ glb, float* __restrict__ z,
                       int outStride, int outOff) {
    __shared__ float g1[64]; __shared__ float red[64];
    int b = blockIdx.x, j = threadIdx.x;               // 64 threads
    float a = gb1[j];
    for (int d = 0; d < 11; ++d) a += g[b * 11 + d] * gw1[j * 11 + d];
    g1[j] = a; __syncthreads();
    float a2 = gb2[j];
    for (int d = 0; d < 64; ++d) a2 += g1[d] * gw2[j * 64 + d];
    float s = g1[j] + a2;
    red[j] = s; __syncthreads();
    for (int o = 32; o > 0; o >>= 1) { if (j < o) red[j] += red[j + o]; __syncthreads(); }
    float mu = red[0] / 64.0f; __syncthreads();
    red[j] = (s - mu) * (s - mu); __syncthreads();
    for (int o = 32; o > 0; o >>= 1) { if (j < o) red[j] += red[j + o]; __syncthreads(); }
    float inv = rsqrtf(red[0] / 64.0f + EPS_);
    z[(size_t)b * outStride + outOff + j] = (s - mu) * inv * glg[j] + glb[j];
}

// -------------------- small FC (+ optional BN + ReLU) ------------------------
__global__ void k_fc(const float* __restrict__ z, const float* __restrict__ w,
                     const float* __restrict__ bias,
                     const float* __restrict__ bng, const float* __restrict__ bnb,
                     const float* __restrict__ bnm, const float* __restrict__ bnv,
                     float* __restrict__ out, int K, int N, int doBnRelu) {
    int idx = blockIdx.x * blockDim.x + threadIdx.x;
    if (idx >= B_ * N) return;
    int b = idx / N, n = idx % N;
    float a = bias[n];
    for (int k = 0; k < K; ++k) a += z[(size_t)b * K + k] * w[(size_t)n * K + k];
    if (doBnRelu) {
        float s = bng[n] * rsqrtf(bnv[n] + EPS_);
        a = fmaxf((a - bnm[n]) * s + bnb[n], 0.0f);
    }
    out[(size_t)b * N + n] = a;
}

// ===========================================================================
extern "C" void kernel_launch(void* const* d_in, const int* in_sizes, int n_in,
                              void* d_out, int out_size, void* d_ws, size_t ws_size,
                              hipStream_t stream)
{
    (void)in_sizes; (void)n_in; (void)out_size; (void)ws_size;
    #define F(i) ((const float*)d_in[i])
    const float* x    = F(0);
    const float* gin  = F(1);
    const int*   lens = (const int*)d_in[2];

    char* ws = (char*)d_ws;
    float*    H64  = (float*)(ws);                       // f32 [B,T,64]   16 MB
    float*    H128 = (float*)(ws + (16ull  << 20));      // f32 [B,T,128]  32 MB
    float*    CAT0 = (float*)(ws + (48ull  << 20));      // f32 [B,T,256]  64 MB (later: LN out)
    float*    CAT1 = (float*)(ws + (112ull << 20));      // f32 [B,T,256]  64 MB
    _Float16* XF16 = (_Float16*)(ws + (176ull << 20));   // f16 [B,T,256]  32 MB
    float*    XG   = (float*)(ws + (208ull << 20));      // f32 [B,T,512] 128 MB
    float*    SCAN = (float*)(ws + (336ull << 20));      // f32 [B,T,128]  32 MB (later: t1)
    char*     WREG = ws + (368ull << 20);                // f16 weights
    char*     SM   = ws + (372ull << 20);                // small buffers

    _Float16* WIH0F = (_Float16*)(WREG);
    _Float16* WIH0B = (_Float16*)(WREG + (64   << 10));
    _Float16* WHH0F = (_Float16*)(WREG + (128  << 10));
    _Float16* WHH0B = (_Float16*)(WREG + (256  << 10));
    _Float16* WIH1F = (_Float16*)(WREG + (384  << 10));
    _Float16* WIH1B = (_Float16*)(WREG + (640  << 10));
    _Float16* WHH1F = (_Float16*)(WREG + (896  << 10));
    _Float16* WHH1B = (_Float16*)(WREG + (1024 << 10));
    _Float16* W1H0  = (_Float16*)(WREG + (1152 << 10));

    float* scores = (float*)(SM);
    float* aw     = (float*)(SM + (256 << 10));
    float* zbuf   = (float*)(SM + (512 << 10));          // [32,320] = ctx(256) | go(64)
    float* z1     = (float*)(SM + (560 << 10));          // [32,128]
    float* z2     = (float*)(SM + (576 << 10));          // [32,64]

    const int M = B_ * T_;                               // 65536
    const size_t smemScan = 8192 + 16384 + 65536 + 131072; // 216 KB

    auto cgrid = [](int n) { return (n + 255) / 256; };

    // ---- fp32 -> fp16 weight staging for WMMA ----
    k_cast_w16<<<cgrid(32768),  256, 0, stream>>>(F(21), WIH0F, 32768);
    k_cast_w16<<<cgrid(32768),  256, 0, stream>>>(F(24), WIH0B, 32768);
    k_cast_w16<<<cgrid(65536),  256, 0, stream>>>(F(22), WHH0F, 65536);
    k_cast_w16<<<cgrid(65536),  256, 0, stream>>>(F(25), WHH0B, 65536);
    k_cast_w16<<<cgrid(131072), 256, 0, stream>>>(F(27), WIH1F, 131072);
    k_cast_w16<<<cgrid(131072), 256, 0, stream>>>(F(30), WIH1B, 131072);
    k_cast_w16<<<cgrid(65536),  256, 0, stream>>>(F(28), WHH1F, 65536);
    k_cast_w16<<<cgrid(65536),  256, 0, stream>>>(F(31), WHH1B, 65536);
    k_cast_w16<<<cgrid(16384),  256, 0, stream>>>(F(35), W1H0,  16384); // attn w1 head 0

    // ---- conv stack: 11 -> 64 -> 128 -> 64 ----
    k_conv_bn_relu<<<cgrid(M * 64),  256, 0, stream>>>(x,    F(3),  F(4),  F(5),  F(6),  F(7),  F(8),  H64,  11,  64);
    k_conv_bn_relu<<<cgrid(M * 128), 256, 0, stream>>>(H64,  F(9),  F(10), F(11), F(12), F(13), F(14), H128, 64, 128);
    k_conv_bn_relu<<<cgrid(M * 64),  256, 0, stream>>>(H128, F(15), F(16), F(17), F(18), F(19), F(20), H64, 128,  64);

    // ---- BiLSTM layer 0 ----
    // forward
    k_cast_act<<<cgrid(M * 64), 256, 0, stream>>>(H64, XF16, lens, 64, 1);       // mask applied here
    k_gemm_bias<<<((M/16)*(512/16) + 3)/4, 128, 0, stream>>>(XF16, WIH0F, F(23), XG, M, 512, 64);
    k_lstm_scan<<<1, 512, smemScan, stream>>>(XG, WHH0F, CAT0, lens, 256, 0, 1);
    // backward
    k_revcast<<<cgrid(M * 64), 256, 0, stream>>>(H64, XF16, lens, 64);
    k_gemm_bias<<<((M/16)*(512/16) + 3)/4, 128, 0, stream>>>(XF16, WIH0B, F(26), XG, M, 512, 64);
    k_lstm_scan<<<1, 512, smemScan, stream>>>(XG, WHH0B, SCAN, lens, 128, 0, 0);
    k_rev_store<<<cgrid(M * 128), 256, 0, stream>>>(SCAN, CAT0, lens, 128, 256, 128);

    // ---- BiLSTM layer 1 ----
    k_cast_act<<<cgrid(M * 256), 256, 0, stream>>>(CAT0, XF16, lens, 256, 0);    // already zero-padded
    k_gemm_bias<<<((M/16)*(512/16) + 3)/4, 128, 0, stream>>>(XF16, WIH1F, F(29), XG, M, 512, 256);
    k_lstm_scan<<<1, 512, smemScan, stream>>>(XG, WHH1F, CAT1, lens, 256, 0, 1);
    k_revcast<<<cgrid(M * 256), 256, 0, stream>>>(CAT0, XF16, lens, 256);
    k_gemm_bias<<<((M/16)*(512/16) + 3)/4, 128, 0, stream>>>(XF16, WIH1B, F(32), XG, M, 512, 256);
    k_lstm_scan<<<1, 512, smemScan, stream>>>(XG, WHH1B, SCAN, lens, 128, 0, 0);
    k_rev_store<<<cgrid(M * 128), 256, 0, stream>>>(SCAN, CAT1, lens, 128, 256, 128);

    // ---- LayerNorm (CAT1 -> CAT0 region, now "nm") ----
    k_layernorm<<<M / 8, 256, 0, stream>>>(CAT1, F(33), F(34), CAT0, 256);

    // ---- attention head 0: t1 = nm @ w1[0]^T + b1[0]; scores; softmax_T; ctx ----
    k_cast_act<<<cgrid(M * 256), 256, 0, stream>>>(CAT0, XF16, lens, 256, 0);
    k_gemm_bias<<<((M/16)*(64/16) + 3)/4, 128, 0, stream>>>(XF16, W1H0, F(36), SCAN, M, 64, 256);
    k_scores<<<M / 256, 256, 0, stream>>>(SCAN, F(37), F(38), scores);
    k_softmax_T<<<B_, 256, 0, stream>>>(scores, aw);
    k_ctx<<<B_, 256, 0, stream>>>(aw, CAT0, zbuf, 320);                          // seq_enc -> z[:, 0:256]

    // ---- global branch -> z[:, 256:320] ----
    k_glob<<<B_, 64, 0, stream>>>(gin, F(39), F(40), F(41), F(42), F(43), F(44), zbuf, 320, 256);

    // ---- classifier ----
    k_fc<<<cgrid(B_ * 128), 256, 0, stream>>>(zbuf, F(45), F(46), F(47), F(48), F(49), F(50), z1, 320, 128, 1);
    k_fc<<<cgrid(B_ * 64),  256, 0, stream>>>(z1,   F(51), F(52), F(53), F(54), F(55), F(56), z2, 128,  64, 1);
    k_fc<<<1, 64, 0, stream>>>(z2, F(57), F(58), nullptr, nullptr, nullptr, nullptr,
                               (float*)d_out, 64, 2, 0);
    #undef F
}
